// NonLocalBlock2d_46162308497530
// MI455X (gfx1250) — matrix operand order repaired and checked by reading, not compiled, not run on hardware
//
#include <hip/hip_runtime.h>

typedef __attribute__((ext_vector_type(16))) __bf16 v16bf;
typedef __attribute__((ext_vector_type(8)))  __bf16 v8bf;
typedef __attribute__((ext_vector_type(8)))  float  v8f;
typedef int v4i_ __attribute__((vector_size(16)));   // matches builtin pointee

constexpr int   B_    = 8;
constexpr int   C_    = 512;
constexpr int   CI    = 256;
constexpr int   N_    = 4096;          // 64*64
constexpr float EPS_  = 1e-5f;
constexpr float SCALE = 0.0625f;       // 1/sqrt(256)
constexpr float LOG2E = 1.4426950408889634f;
constexpr float SCL2  = SCALE * LOG2E; // logits scaled into log2 domain

// ---- workspace layout (bytes) ----
constexpr size_t WT_OFF = 0;
constexpr size_t WP_OFF = WT_OFF + (size_t)CI * C_ * 2;
constexpr size_t WG_OFF = WP_OFF + (size_t)CI * C_ * 2;
constexpr size_t WZ_OFF = WG_OFF + (size_t)CI * C_ * 2;
constexpr size_t BT_OFF = WZ_OFF + (size_t)C_ * CI * 2;
constexpr size_t BP_OFF = BT_OFF + (size_t)CI * 4;
constexpr size_t BG_OFF = BP_OFF + (size_t)CI * 4;
constexpr size_t BZ_OFF = BG_OFF + (size_t)CI * 4;
constexpr size_t ACT_OFF = (BZ_OFF + (size_t)C_ * 4 + 255) & ~(size_t)255;
constexpr size_t TH_OFF = ACT_OFF;                               // thT [B][N][CI] bf16
constexpr size_t PH_OFF = TH_OFF + (size_t)B_ * N_ * CI * 2;     // phT [B][N][CI] bf16
constexpr size_t GX_OFF = PH_OFF + (size_t)B_ * N_ * CI * 2;     // g   [B][CI][N] bf16
constexpr size_t YT_OFF = GX_OFF + (size_t)B_ * CI * N_ * 2;     // yT  [B][N][CI] bf16

#if defined(__gfx1250__) && __has_builtin(__builtin_amdgcn_global_load_async_to_lds_b128)
#define HAVE_ASYNC_LDS 1
#else
#define HAVE_ASYNC_LDS 0
#endif

__device__ __forceinline__ void copy16_to_lds(__bf16* lds, const __bf16* g) {
#if HAVE_ASYNC_LDS
  __builtin_amdgcn_global_load_async_to_lds_b128(
      (__attribute__((address_space(1))) v4i_*)(g),
      (__attribute__((address_space(3))) v4i_*)(uint32_t)(uintptr_t)(lds),
      0, 0);
#else
  *(v8bf*)lds = *(const v8bf*)g;
#endif
}

__device__ __forceinline__ void async_fence_block() {
#if HAVE_ASYNC_LDS
#if __has_builtin(__builtin_amdgcn_s_wait_asynccnt)
  __builtin_amdgcn_s_wait_asynccnt(0);
#else
  asm volatile("s_wait_asynccnt 0x0" ::: "memory");
#endif
#endif
  __syncthreads();
}

__device__ __forceinline__ v16bf cat16(v8bf lo, v8bf hi) {
  v16bf r;
#pragma unroll
  for (int i = 0; i < 8; ++i) { r[i] = lo[i]; r[i + 8] = hi[i]; }
  return r;
}

__device__ __forceinline__ v8f wmma_bf16(v16bf a, v16bf b, v8f c) {
  return __builtin_amdgcn_wmma_f32_16x16x32_bf16(false, a, false, b, (short)0, c,
                                                 false, false);
}

// ---------------------------------------------------------------------------
// Kernel A: fold BN into conv weights, cast to bf16
// ---------------------------------------------------------------------------
__global__ __launch_bounds__(256) void fold_weights(
    const float* __restrict__ tw, const float* __restrict__ tb,
    const float* __restrict__ tg, const float* __restrict__ tbb,
    const float* __restrict__ tm, const float* __restrict__ tv,
    const float* __restrict__ pw, const float* __restrict__ pb,
    const float* __restrict__ pg, const float* __restrict__ pbb,
    const float* __restrict__ pm, const float* __restrict__ pv,
    const float* __restrict__ gw, const float* __restrict__ gb,
    const float* __restrict__ zw, const float* __restrict__ zb,
    const float* __restrict__ zg, const float* __restrict__ zbb,
    const float* __restrict__ zm, const float* __restrict__ zv,
    char* __restrict__ ws) {
  int i = blockIdx.x * blockDim.x + threadIdx.x;
  __bf16* wT = (__bf16*)(ws + WT_OFF);
  __bf16* wP = (__bf16*)(ws + WP_OFF);
  __bf16* wG = (__bf16*)(ws + WG_OFF);
  __bf16* wZ = (__bf16*)(ws + WZ_OFF);
  float*  bT = (float*)(ws + BT_OFF);
  float*  bP = (float*)(ws + BP_OFF);
  float*  bG = (float*)(ws + BG_OFF);
  float*  bZ = (float*)(ws + BZ_OFF);
  if (i < CI * C_) {
    int o = i / C_;
    float st = tg[o] * rsqrtf(tv[o] + EPS_);
    wT[i] = (__bf16)(tw[i] * st);
    float sp = pg[o] * rsqrtf(pv[o] + EPS_);
    wP[i] = (__bf16)(pw[i] * sp);
    wG[i] = (__bf16)gw[i];
    int o2 = i / CI;
    float sz = zg[o2] * rsqrtf(zv[o2] + EPS_);
    wZ[i] = (__bf16)(zw[i] * sz);
  }
  if (i < CI) {
    float st = tg[i] * rsqrtf(tv[i] + EPS_);
    bT[i] = (tb[i] - tm[i]) * st + tbb[i];
    float sp = pg[i] * rsqrtf(pv[i] + EPS_);
    bP[i] = (pb[i] - pm[i]) * sp + pbb[i];
    bG[i] = gb[i];
  }
  if (i < C_) {
    float sz = zg[i] * rsqrtf(zv[i] + EPS_);
    bZ[i] = (zb[i] - zm[i]) * sz + zbb[i];
  }
}

// ---------------------------------------------------------------------------
// Kernel B: projections P = W_eff @ x  (bf16 WMMA, f32 accum)
// grid: (N/128, CI/128, 3*B).  theta/phi stored transposed [n][ci]; g [ci][n].
// ---------------------------------------------------------------------------
__global__ __launch_bounds__(256) void proj_gemm(const float* __restrict__ x,
                                                 char* __restrict__ ws) {
  __shared__ __attribute__((aligned(16))) __bf16 sB[128][40];  // x^T chunk [n][c]
  int z = blockIdx.z;
  int proj = z % 3, b = z / 3;
  const __bf16* W =
      (const __bf16*)(ws + (proj == 0 ? WT_OFF : proj == 1 ? WP_OFF : WG_OFF));
  const float* bias =
      (const float*)(ws + (proj == 0 ? BT_OFF : proj == 1 ? BP_OFF : BG_OFF));
  const float* xb = x + (size_t)b * C_ * N_;
  int nBase = blockIdx.x * 128;
  int mBase = blockIdx.y * 128;
  int tid = threadIdx.x;
  int wave = tid >> 5, lane = tid & 31;
  int lrow = lane & 15, half = lane >> 4;
  int waveM = wave >> 2, waveN = wave & 3;   // 2 x 4 waves, 64x32 per wave

  v8f acc[4][2];
  v8f zero = {0.f, 0.f, 0.f, 0.f, 0.f, 0.f, 0.f, 0.f};
#pragma unroll
  for (int i = 0; i < 4; ++i)
#pragma unroll
    for (int j = 0; j < 2; ++j) acc[i][j] = zero;

  for (int kk = 0; kk < C_; kk += 32) {
    __syncthreads();
#pragma unroll
    for (int it = 0; it < 16; ++it) {
      int idx = tid + it * 256;
      int c = idx >> 7, n = idx & 127;
      sB[n][c] = (__bf16)xb[(size_t)(kk + c) * N_ + nBase + n];
    }
    __syncthreads();

    v16bf Af[4];
#pragma unroll
    for (int i = 0; i < 4; ++i) {
      const __bf16* ar =
          W + (size_t)(mBase + waveM * 64 + i * 16 + lrow) * C_ + kk;
      Af[i] = cat16(*(const v8bf*)(ar + half * 8),
                    *(const v8bf*)(ar + 16 + half * 8));
    }
    v16bf Bf[2];
#pragma unroll
    for (int j = 0; j < 2; ++j) {
      const __bf16* br = &sB[waveN * 32 + j * 16 + lrow][half * 8];
      Bf[j] = cat16(*(const v8bf*)br, *(const v8bf*)(br + 16));
    }
#pragma unroll
    for (int i = 0; i < 4; ++i)
#pragma unroll
      for (int j = 0; j < 2; ++j) acc[i][j] = wmma_bf16(Af[i], Bf[j], acc[i][j]);
  }

  __bf16* outT =
      (__bf16*)(ws + (proj == 0 ? TH_OFF : PH_OFF)) + (size_t)b * N_ * CI;
  __bf16* outG = (__bf16*)(ws + GX_OFF) + (size_t)b * CI * N_;
#pragma unroll
  for (int i = 0; i < 4; ++i) {
    int rowBase = mBase + waveM * 64 + i * 16 + half * 8;  // ci
#pragma unroll
    for (int j = 0; j < 2; ++j) {
      int col = nBase + waveN * 32 + j * 16 + lrow;        // n
      if (proj < 2) {
        v8bf pk;
#pragma unroll
        for (int r = 0; r < 8; ++r)
          pk[r] = (__bf16)(acc[i][j][r] + bias[rowBase + r]);
        *(v8bf*)(outT + (size_t)col * CI + rowBase) = pk;  // [n][ci]
      } else {
#pragma unroll
        for (int r = 0; r < 8; ++r)
          outG[(size_t)(rowBase + r) * N_ + col] =
              (__bf16)(acc[i][j][r] + bias[rowBase + r]); // [ci][n]
      }
    }
  }
}

// ---------------------------------------------------------------------------
// Kernel C: flash attention.  grid (N/128, B); wave owns 16 query rows.
// Key/value tiles (Tk=32) staged block-wide in LDS via async-to-LDS;
// online softmax in base-2 (native v_exp_f32).
// ---------------------------------------------------------------------------
__global__ __launch_bounds__(256) void flash_attn(char* __restrict__ ws) {
  __shared__ __attribute__((aligned(16))) __bf16 sPh[32][264];   // [m][ci]
  __shared__ __attribute__((aligned(16))) __bf16 sG[256][40];    // [ci][m]
  __shared__ __attribute__((aligned(16))) __bf16 sP[8][16][40];  // per-wave P
  int b = blockIdx.y;
  const __bf16* thp = (const __bf16*)(ws + TH_OFF) + (size_t)b * N_ * CI;
  const __bf16* php = (const __bf16*)(ws + PH_OFF) + (size_t)b * N_ * CI;
  const __bf16* gp  = (const __bf16*)(ws + GX_OFF) + (size_t)b * CI * N_;
  __bf16* yp = (__bf16*)(ws + YT_OFF) + (size_t)b * N_ * CI;

  int tid = threadIdx.x, wave = tid >> 5, lane = tid & 31;
  int lrow = lane & 15, half = lane >> 4;
  int qBase = blockIdx.x * 128 + wave * 16;

  // preload query A-fragments (16 rows x 256 c = 8 chunks)
  v16bf Aq[8];
#pragma unroll
  for (int k = 0; k < 8; ++k) {
    const __bf16* ar = thp + (size_t)(qBase + lrow) * CI + k * 32;
    Aq[k] = cat16(*(const v8bf*)(ar + half * 8),
                  *(const v8bf*)(ar + 16 + half * 8));
  }

  v8f zero = {0.f, 0.f, 0.f, 0.f, 0.f, 0.f, 0.f, 0.f};
  float mrow[8], lsum[8];
  v8f acc[16];  // 16 q-rows x 256 ci
#pragma unroll
  for (int r = 0; r < 8; ++r) { mrow[r] = -1e30f; lsum[r] = 0.f; }
#pragma unroll
  for (int t = 0; t < 16; ++t) acc[t] = zero;

  for (int kt = 0; kt < N_ / 32; ++kt) {
    int mB = kt * 32;
    // ---- stage key tile [32 m][256 ci] and value tile [256 ci][32 m] ----
#pragma unroll
    for (int it = 0; it < 4; ++it) {
      int ch = tid + it * 256;                   // 1024 x 16B chunks
      int m = ch >> 5, cc = (ch & 31) * 8;       // sPh
      copy16_to_lds(&sPh[m][cc], php + (size_t)(mB + m) * CI + cc);
      int ci = ch >> 2, mm = (ch & 3) * 8;       // sG
      copy16_to_lds(&sG[ci][mm], gp + (size_t)ci * N_ + mB + mm);
    }
    if (kt + 1 < N_ / 32) {                      // prefetch next tiles
      int ch = tid;
      __builtin_prefetch(php + (size_t)(mB + 32 + (ch >> 3)) * CI + (ch & 7) * 32, 0, 0);
      __builtin_prefetch(gp + (size_t)ch * N_ + mB + 32, 0, 0);
    }
    async_fence_block();

    // ---- S = th^T @ ph : 2 key sub-tiles of 16 ----
    v8f Sf[2];
#pragma unroll
    for (int j = 0; j < 2; ++j) {
      v8f s = zero;
#pragma unroll
      for (int k = 0; k < 8; ++k) {
        const __bf16* br = &sPh[j * 16 + lrow][k * 32 + half * 8];
        v16bf Bf = cat16(*(const v8bf*)br, *(const v8bf*)(br + 16));
        s = wmma_bf16(Aq[k], Bf, s);
      }
#pragma unroll
      for (int r = 0; r < 8; ++r) s[r] *= SCL2;  // log2-domain logits
      Sf[j] = s;
    }
    // ---- online softmax (rows striped r + 8*half; cols = lanes in half) ----
    float fac[8];
#pragma unroll
    for (int r = 0; r < 8; ++r) {
      float tm = fmaxf(Sf[0][r], Sf[1][r]);
      tm = fmaxf(tm, __shfl_xor(tm, 1, 32));
      tm = fmaxf(tm, __shfl_xor(tm, 2, 32));
      tm = fmaxf(tm, __shfl_xor(tm, 4, 32));
      tm = fmaxf(tm, __shfl_xor(tm, 8, 32));
      float mn = fmaxf(mrow[r], tm);
      fac[r] = exp2f(mrow[r] - mn);
      mrow[r] = mn;
    }
#pragma unroll
    for (int r = 0; r < 8; ++r) {
      float rs = 0.f;
#pragma unroll
      for (int j = 0; j < 2; ++j) {
        float p = exp2f(Sf[j][r] - mrow[r]);
        Sf[j][r] = p;
        rs += p;
      }
      rs += __shfl_xor(rs, 1, 32);
      rs += __shfl_xor(rs, 2, 32);
      rs += __shfl_xor(rs, 4, 32);
      rs += __shfl_xor(rs, 8, 32);
      lsum[r] = lsum[r] * fac[r] + rs;
    }
#pragma unroll
    for (int t = 0; t < 16; ++t)
#pragma unroll
      for (int r = 0; r < 8; ++r) acc[t][r] *= fac[r];
    // ---- P (C-layout) -> per-wave LDS -> A-layout fragment ----
#pragma unroll
    for (int j = 0; j < 2; ++j)
#pragma unroll
      for (int r = 0; r < 8; ++r)
        sP[wave][r + half * 8][j * 16 + lrow] = (__bf16)Sf[j][r];
    // ---- O += P @ g^T ----
    {
      const __bf16* pr = &sP[wave][lrow][half * 8];
      v16bf Ap = cat16(*(const v8bf*)pr, *(const v8bf*)(pr + 16));
#pragma unroll
      for (int t = 0; t < 16; ++t) {
        const __bf16* br = &sG[t * 16 + lrow][half * 8];
        v16bf Bf = cat16(*(const v8bf*)br, *(const v8bf*)(br + 16));
        acc[t] = wmma_bf16(Ap, Bf, acc[t]);
      }
    }
    __syncthreads();  // protect sPh/sG before next stage
  }
  // ---- normalize and store y^T [n][ci] bf16 ----
#pragma unroll
  for (int r = 0; r < 8; ++r) lsum[r] = 1.0f / lsum[r];
#pragma unroll
  for (int t = 0; t < 16; ++t)
#pragma unroll
    for (int r = 0; r < 8; ++r)
      yp[(size_t)(qBase + r + half * 8) * CI + t * 16 + lrow] =
          (__bf16)(acc[t][r] * lsum[r]);
}

// ---------------------------------------------------------------------------
// Kernel D: out = Wz_eff @ y + bias + x   (residual), f32 output
// grid: (N/128, C/128, B). All fragments straight from global (contiguous).
// ---------------------------------------------------------------------------
__global__ __launch_bounds__(256) void out_gemm(const float* __restrict__ x,
                                                float* __restrict__ out,
                                                const char* __restrict__ ws) {
  int b = blockIdx.z;
  const __bf16* Wz = (const __bf16*)(ws + WZ_OFF);
  const float*  bz = (const float*)(ws + BZ_OFF);
  const __bf16* yt = (const __bf16*)(ws + YT_OFF) + (size_t)b * N_ * CI;
  int nBase = blockIdx.x * 128, mBase = blockIdx.y * 128;
  int tid = threadIdx.x, wave = tid >> 5, lane = tid & 31;
  int lrow = lane & 15, half = lane >> 4;
  int waveM = wave >> 2, waveN = wave & 3;

  v8f zero = {0.f, 0.f, 0.f, 0.f, 0.f, 0.f, 0.f, 0.f};
  v8f acc[4][2];
#pragma unroll
  for (int i = 0; i < 4; ++i)
#pragma unroll
    for (int j = 0; j < 2; ++j) acc[i][j] = zero;

  for (int kk = 0; kk < CI; kk += 32) {
    v16bf Af[4];
#pragma unroll
    for (int i = 0; i < 4; ++i) {
      const __bf16* ar =
          Wz + (size_t)(mBase + waveM * 64 + i * 16 + lrow) * CI + kk;
      Af[i] = cat16(*(const v8bf*)(ar + half * 8),
                    *(const v8bf*)(ar + 16 + half * 8));
    }
    v16bf Bf[2];
#pragma unroll
    for (int j = 0; j < 2; ++j) {
      const __bf16* br =
          yt + (size_t)(nBase + waveN * 32 + j * 16 + lrow) * CI + kk + half * 8;
      Bf[j] = cat16(*(const v8bf*)br, *(const v8bf*)(br + 16));
    }
#pragma unroll
    for (int i = 0; i < 4; ++i)
#pragma unroll
      for (int j = 0; j < 2; ++j) acc[i][j] = wmma_bf16(Af[i], Bf[j], acc[i][j]);
  }

  const float* xb = x + (size_t)b * C_ * N_;
  float* ob = out + (size_t)b * C_ * N_;
#pragma unroll
  for (int i = 0; i < 4; ++i) {
    int rowBase = mBase + waveM * 64 + i * 16 + half * 8;
#pragma unroll
    for (int j = 0; j < 2; ++j) {
      int col = nBase + waveN * 32 + j * 16 + lrow;
#pragma unroll
      for (int r = 0; r < 8; ++r) {
        int row = rowBase + r;
        size_t a = (size_t)row * N_ + col;
        ob[a] = acc[i][j][r] + bz[row] + xb[a];
      }
    }
  }
}

// ---------------------------------------------------------------------------
extern "C" void kernel_launch(void* const* d_in, const int* in_sizes, int n_in,
                              void* d_out, int out_size, void* d_ws,
                              size_t ws_size, hipStream_t stream) {
  (void)in_sizes; (void)n_in; (void)out_size; (void)ws_size;
  const float* x = (const float*)d_in[0];
  char* ws = (char*)d_ws;

  fold_weights<<<dim3((CI * C_ + 255) / 256), 256, 0, stream>>>(
      (const float*)d_in[1], (const float*)d_in[2], (const float*)d_in[3],
      (const float*)d_in[4], (const float*)d_in[5], (const float*)d_in[6],
      (const float*)d_in[7], (const float*)d_in[8], (const float*)d_in[9],
      (const float*)d_in[10], (const float*)d_in[11], (const float*)d_in[12],
      (const float*)d_in[13], (const float*)d_in[14], (const float*)d_in[15],
      (const float*)d_in[16], (const float*)d_in[17], (const float*)d_in[18],
      (const float*)d_in[19], (const float*)d_in[20], ws);

  proj_gemm<<<dim3(N_ / 128, CI / 128, 3 * B_), 256, 0, stream>>>(x, ws);
  flash_attn<<<dim3(N_ / 128, B_), 256, 0, stream>>>(ws);
  out_gemm<<<dim3(N_ / 128, C_ / 128, B_), 256, 0, stream>>>(
      x, (float*)d_out, ws);
}
